// MultiHeadAttention_34823594836323
// MI455X (gfx1250) — compile-verified, matrix operand order checked by reading, not run on hardware
//
#include <hip/hip_runtime.h>
#include <hip/hip_bf16.h>

typedef __attribute__((ext_vector_type(16))) _Float16 v16h;
typedef __attribute__((ext_vector_type(8)))  _Float16 v8h;
typedef __attribute__((ext_vector_type(8)))  float    v8f;

#define WMMA_F32_F16(A, Bm, C) \
  __builtin_amdgcn_wmma_f32_16x16x32_f16(false, (A), false, (Bm), (short)0, (C), false, false)

namespace {
constexpr int B_   = 4;
constexpr int S_   = 2048;
constexpr int D_   = 1024;
constexpr int H_   = 16;
constexpr int DK_  = 64;
constexpr int DV_  = 64;
constexpr int DOUT_= 1024;
constexpr size_t HEAD_ELEMS = (size_t)B_ * H_ * S_ * DK_;   // per qh/kh/vhT buffer
}

// Build a 16x32-f16 WMMA operand fragment from two contiguous 8-half (16B) runs.
__device__ __forceinline__ v16h mk16(const _Float16* lo_p, const _Float16* hi_p) {
  v8h lo = *(const v8h*)lo_p;
  v8h hi = *(const v8h*)hi_p;
  v16h r;
#pragma unroll
  for (int i = 0; i < 8; ++i) { r[i] = lo[i]; r[i + 8] = hi[i]; }
  return r;
}

// ---------------------------------------------------------------------------
// Kernel 1: per-head QKV projections.
//   grid = (S/128, B*H, 3) ; block = 256 (8 waves)
//   which==0: qh[b][h][s][e] = q·Wq + bq   (row-major f16)
//   which==1: kh[b][h][s][e] = k·Wk + bk   (row-major f16, = B-layout [N=sk][K=dk])
//   which==2: vhT[b][h][e][s] = v·Wv + bv  (transposed, = B-layout [N=dv][K=sk])
// ---------------------------------------------------------------------------
__global__ __launch_bounds__(256) void mha_proj_kernel(
    const float* __restrict__ q, const float* __restrict__ k, const float* __restrict__ v,
    const float* __restrict__ Wq, const float* __restrict__ Wk, const float* __restrict__ Wv,
    const float* __restrict__ bq, const float* __restrict__ bk, const float* __restrict__ bv,
    _Float16* __restrict__ qh, _Float16* __restrict__ kh, _Float16* __restrict__ vhT)
{
  __shared__ _Float16 As[128][40];  // [m][k] f16, padded
  __shared__ _Float16 Bs[64][40];   // [n][k] f16 (transposed weight tile)

  const int which = blockIdx.z;
  const int bh = blockIdx.y;
  const int b  = bh / H_, h = bh % H_;
  const int s0 = blockIdx.x * 128;
  const int tid  = threadIdx.x;
  const int lane = tid & 31, wave = tid >> 5;
  const int lidx = lane & 15, khalf = lane >> 4;

  const float* X    = (which == 0) ? q  : (which == 1) ? k  : v;
  const float* W    = ((which == 0) ? Wq : (which == 1) ? Wk : Wv) + (size_t)h * D_ * DK_;
  const float* bias = ((which == 0) ? bq : (which == 1) ? bk : bv) + (size_t)h * DK_;

  v8f acc[4];
#pragma unroll
  for (int i = 0; i < 4; ++i) acc[i] = v8f{0.f,0.f,0.f,0.f,0.f,0.f,0.f,0.f};

  for (int k0 = 0; k0 < D_; k0 += 32) {
    // --- stage A tile: X[s0..s0+127][k0..k0+31] f32 -> f16 LDS ---
    {
      const int row = tid >> 1, kc = (tid & 1) * 16;
      const float* src = X + ((size_t)(b * S_ + s0 + row)) * D_ + k0 + kc;
      float4 f0 = *(const float4*)(src + 0);
      float4 f1 = *(const float4*)(src + 4);
      float4 f2 = *(const float4*)(src + 8);
      float4 f3 = *(const float4*)(src + 12);
      v8h h0 = {(_Float16)f0.x,(_Float16)f0.y,(_Float16)f0.z,(_Float16)f0.w,
                (_Float16)f1.x,(_Float16)f1.y,(_Float16)f1.z,(_Float16)f1.w};
      v8h h1 = {(_Float16)f2.x,(_Float16)f2.y,(_Float16)f2.z,(_Float16)f2.w,
                (_Float16)f3.x,(_Float16)f3.y,(_Float16)f3.z,(_Float16)f3.w};
      *(v8h*)&As[row][kc]     = h0;
      *(v8h*)&As[row][kc + 8] = h1;
    }
    // --- stage B tile: W[k0+kk][n] -> Bs[n][kk] (transpose) ---
    {
      const int kk = tid >> 3, nn = (tid & 7) * 8;
      const float* wsrc = W + (size_t)(k0 + kk) * DK_ + nn;
      float4 f0 = *(const float4*)(wsrc);
      float4 f1 = *(const float4*)(wsrc + 4);
      Bs[nn + 0][kk] = (_Float16)f0.x; Bs[nn + 1][kk] = (_Float16)f0.y;
      Bs[nn + 2][kk] = (_Float16)f0.z; Bs[nn + 3][kk] = (_Float16)f0.w;
      Bs[nn + 4][kk] = (_Float16)f1.x; Bs[nn + 5][kk] = (_Float16)f1.y;
      Bs[nn + 6][kk] = (_Float16)f1.z; Bs[nn + 7][kk] = (_Float16)f1.w;
    }
    __syncthreads();

    const int arow = wave * 16 + lidx;
    v16h a = mk16(&As[arow][khalf * 8], &As[arow][16 + khalf * 8]);
#pragma unroll
    for (int nt = 0; nt < 4; ++nt) {
      const int brow = nt * 16 + lidx;
      v16h bm = mk16(&Bs[brow][khalf * 8], &Bs[brow][16 + khalf * 8]);
      acc[nt] = WMMA_F32_F16(a, bm, acc[nt]);
    }
    __syncthreads();
  }

  // --- epilogue: bias + store ---
#pragma unroll
  for (int nt = 0; nt < 4; ++nt) {
    const int n = nt * 16 + lidx;
    const float bb = bias[n];
    if (which < 2) {
      _Float16* dst = ((which == 0) ? qh : kh) + (size_t)(b * H_ + h) * S_ * DK_;
#pragma unroll
      for (int r = 0; r < 8; ++r) {
        const int s = s0 + wave * 16 + r + 8 * khalf;
        dst[(size_t)s * DK_ + n] = (_Float16)(acc[nt][r] + bb);
      }
    } else {
      v8h pk;
#pragma unroll
      for (int r = 0; r < 8; ++r) pk[r] = (_Float16)(acc[nt][r] + bb);
      _Float16* dst = vhT + ((size_t)(b * H_ + h) * DV_ + n) * S_ + s0 + wave * 16 + 8 * khalf;
      *(v8h*)dst = pk;  // 8 consecutive s positions -> one b128 store
    }
  }
}

// ---------------------------------------------------------------------------
// Kernel 2: flash attention per (b, h, 128-row q tile).
//   grid = (S/128, B*H) ; block = 256 (8 waves, each owns 16 query rows)
// ---------------------------------------------------------------------------
__global__ __launch_bounds__(256) void mha_attn_kernel(
    const _Float16* __restrict__ qh, const _Float16* __restrict__ kh,
    const _Float16* __restrict__ vhT, const unsigned char* __restrict__ mask,
    _Float16* __restrict__ concat)
{
  __shared__ _Float16 Ks[32][72];      // [sk][dk]  (B-layout for scores)
  __shared__ _Float16 Vs[64][40];      // [dv][sk]  (B-layout for PV)
  __shared__ _Float16 Ps[8][16][40];   // per-wave C->A relayout scratch

  const int bh = blockIdx.y, b = bh / H_, h = bh % H_;
  const int s0 = blockIdx.x * 128;
  const int tid  = threadIdx.x;
  const int lane = tid & 31, wave = tid >> 5;
  const int lidx = lane & 15, khalf = lane >> 4;

  const size_t headbase = (size_t)(b * H_ + h) * S_ * DK_;
  const size_t vbase    = (size_t)(b * H_ + h) * DV_ * S_;

  // Q fragments for this wave's 16 rows (DK=64 -> two 16x32 A fragments)
  const _Float16* qrow = qh + headbase + (size_t)(s0 + wave * 16 + lidx) * DK_;
  const v16h qa0 = mk16(qrow + khalf * 8,      qrow + 16 + khalf * 8);
  const v16h qa1 = mk16(qrow + 32 + khalf * 8, qrow + 48 + khalf * 8);

  float rmax[8], rsum[8];
  v8f oacc[4];
#pragma unroll
  for (int r = 0; r < 8; ++r) { rmax[r] = -1e30f; rsum[r] = 0.f; }
#pragma unroll
  for (int i = 0; i < 4; ++i) oacc[i] = v8f{0.f,0.f,0.f,0.f,0.f,0.f,0.f,0.f};

  const float scale = 0.125f;  // 1/sqrt(DK)
  const unsigned char* mrow = mask + (size_t)b * S_;
  const v8f zero = v8f{0.f,0.f,0.f,0.f,0.f,0.f,0.f,0.f};

  for (int sk0 = 0; sk0 < S_; sk0 += 32) {
    // cooperative tile loads
    {
      const int skr = tid >> 3, dd = (tid & 7) * 8;
      *(v8h*)&Ks[skr][dd] = *(const v8h*)(kh + headbase + (size_t)(sk0 + skr) * DK_ + dd);
    }
    {
      const int dv = tid >> 2, sc = (tid & 3) * 8;
      *(v8h*)&Vs[dv][sc] = *(const v8h*)(vhT + vbase + (size_t)dv * S_ + sk0 + sc);
    }
    __syncthreads();

    // scores: two 16x16 tiles, K=DK=64 split over 2 WMMAs each
    float p[2][8];
#pragma unroll
    for (int nt = 0; nt < 2; ++nt) {
      const int krow = nt * 16 + lidx;
      v16h kf0 = mk16(&Ks[krow][khalf * 8],      &Ks[krow][16 + khalf * 8]);
      v16h kf1 = mk16(&Ks[krow][32 + khalf * 8], &Ks[krow][48 + khalf * 8]);
      v8f sc2 = WMMA_F32_F16(qa0, kf0, zero);
      sc2 = WMMA_F32_F16(qa1, kf1, sc2);
      const bool mv = mrow[sk0 + nt * 16 + lidx] != 0;  // per-lane key column
#pragma unroll
      for (int r = 0; r < 8; ++r) p[nt][r] = mv ? sc2[r] * scale : -1e30f;
    }

    // online softmax (row reductions across the 16 lanes of each half)
#pragma unroll
    for (int r = 0; r < 8; ++r) {
      float m = fmaxf(p[0][r], p[1][r]);
      m = fmaxf(m, __shfl_xor(m, 1));
      m = fmaxf(m, __shfl_xor(m, 2));
      m = fmaxf(m, __shfl_xor(m, 4));
      m = fmaxf(m, __shfl_xor(m, 8));
      const float nm = fmaxf(rmax[r], m);
      const float cf = __expf(rmax[r] - nm);
      rmax[r] = nm;
      const float p0 = __expf(p[0][r] - nm);
      const float p1 = __expf(p[1][r] - nm);
      p[0][r] = p0; p[1][r] = p1;
      float ts = p0 + p1;
      ts += __shfl_xor(ts, 1);
      ts += __shfl_xor(ts, 2);
      ts += __shfl_xor(ts, 4);
      ts += __shfl_xor(ts, 8);
      rsum[r] = rsum[r] * cf + ts;
#pragma unroll
      for (int vt = 0; vt < 4; ++vt) oacc[vt][r] *= cf;
    }

    // C-layout -> A-layout relayout of P through per-wave LDS scratch
#pragma unroll
    for (int nt = 0; nt < 2; ++nt)
#pragma unroll
      for (int r = 0; r < 8; ++r)
        Ps[wave][r + 8 * khalf][nt * 16 + lidx] = (_Float16)p[nt][r];
    __syncthreads();

    v16h pf = mk16(&Ps[wave][lidx][khalf * 8], &Ps[wave][lidx][16 + khalf * 8]);
#pragma unroll
    for (int vt = 0; vt < 4; ++vt) {
      const int vrow = vt * 16 + lidx;
      v16h vf = mk16(&Vs[vrow][khalf * 8], &Vs[vrow][16 + khalf * 8]);
      oacc[vt] = WMMA_F32_F16(pf, vf, oacc[vt]);
    }
    __syncthreads();
  }

  // normalize + store to concat (B, S, H*DV) as f16
#pragma unroll
  for (int vt = 0; vt < 4; ++vt) {
    const int dv = vt * 16 + lidx;
#pragma unroll
    for (int r = 0; r < 8; ++r) {
      const int s = s0 + wave * 16 + r + 8 * khalf;
      const float denom = fmaxf(rsum[r], 1e-20f);
      concat[((size_t)b * S_ + s) * (H_ * DV_) + h * DV_ + dv] =
          (_Float16)(oacc[vt][r] / denom);
    }
  }
}

// ---------------------------------------------------------------------------
// Kernel 3: output projection  out = concat(8192x1024) * Wo + bo  (f32 out)
//   grid = (B*S/128, DOUT/64) ; block = 256
// ---------------------------------------------------------------------------
__global__ __launch_bounds__(256) void mha_outproj_kernel(
    const _Float16* __restrict__ concat, const float* __restrict__ Wo,
    const float* __restrict__ bo, float* __restrict__ out)
{
  __shared__ _Float16 Bs[64][40];  // Wo tile transposed: [n][k] f16

  const int row0 = blockIdx.x * 128;
  const int n0   = blockIdx.y * 64;
  const int tid  = threadIdx.x;
  const int lane = tid & 31, wave = tid >> 5;
  const int lidx = lane & 15, khalf = lane >> 4;

  v8f acc[4];
#pragma unroll
  for (int i = 0; i < 4; ++i) acc[i] = v8f{0.f,0.f,0.f,0.f,0.f,0.f,0.f,0.f};

  const _Float16* arow = concat + (size_t)(row0 + wave * 16 + lidx) * (H_ * DV_);

  for (int k0 = 0; k0 < H_ * DV_; k0 += 32) {
    {
      const int kk = tid >> 3, nn = (tid & 7) * 8;
      const float* wsrc = Wo + (size_t)(k0 + kk) * DOUT_ + n0 + nn;
      float4 f0 = *(const float4*)(wsrc);
      float4 f1 = *(const float4*)(wsrc + 4);
      Bs[nn + 0][kk] = (_Float16)f0.x; Bs[nn + 1][kk] = (_Float16)f0.y;
      Bs[nn + 2][kk] = (_Float16)f0.z; Bs[nn + 3][kk] = (_Float16)f0.w;
      Bs[nn + 4][kk] = (_Float16)f1.x; Bs[nn + 5][kk] = (_Float16)f1.y;
      Bs[nn + 6][kk] = (_Float16)f1.z; Bs[nn + 7][kk] = (_Float16)f1.w;
    }
    __syncthreads();

    v16h a = mk16(arow + k0 + khalf * 8, arow + k0 + 16 + khalf * 8);
#pragma unroll
    for (int nt = 0; nt < 4; ++nt) {
      const int brow = nt * 16 + lidx;
      v16h bm = mk16(&Bs[brow][khalf * 8], &Bs[brow][16 + khalf * 8]);
      acc[nt] = WMMA_F32_F16(a, bm, acc[nt]);
    }
    __syncthreads();
  }

#pragma unroll
  for (int nt = 0; nt < 4; ++nt) {
    const int n = n0 + nt * 16 + lidx;
    const float bb = bo[n];
#pragma unroll
    for (int r = 0; r < 8; ++r) {
      const int row = row0 + wave * 16 + r + 8 * khalf;
      out[(size_t)row * DOUT_ + n] = acc[nt][r] + bb;
    }
  }
}

// ---------------------------------------------------------------------------
extern "C" void kernel_launch(void* const* d_in, const int* in_sizes, int n_in,
                              void* d_out, int out_size, void* d_ws, size_t ws_size,
                              hipStream_t stream) {
  (void)in_sizes; (void)n_in; (void)out_size; (void)ws_size;

  const float* q  = (const float*)d_in[0];
  const float* k  = (const float*)d_in[1];
  const float* v  = (const float*)d_in[2];
  const unsigned char* mask = (const unsigned char*)d_in[3];
  const float* Wq = (const float*)d_in[4];
  const float* bq = (const float*)d_in[5];
  const float* Wk = (const float*)d_in[6];
  const float* bk = (const float*)d_in[7];
  const float* Wv = (const float*)d_in[8];
  const float* bv = (const float*)d_in[9];
  const float* Wo = (const float*)d_in[10];
  const float* bo = (const float*)d_in[11];
  float* out = (float*)d_out;

  _Float16* ws     = (_Float16*)d_ws;
  _Float16* qh     = ws;
  _Float16* kh     = ws + HEAD_ELEMS;
  _Float16* vhT    = ws + 2 * HEAD_ELEMS;
  _Float16* concat = ws + 3 * HEAD_ELEMS;

  hipLaunchKernelGGL(mha_proj_kernel, dim3(S_ / 128, B_ * H_, 3), dim3(256), 0, stream,
                     q, k, v, Wq, Wk, Wv, bq, bk, bv, qh, kh, vhT);
  hipLaunchKernelGGL(mha_attn_kernel, dim3(S_ / 128, B_ * H_), dim3(256), 0, stream,
                     qh, kh, vhT, mask, concat);
  hipLaunchKernelGGL(mha_outproj_kernel, dim3((B_ * S_) / 128, DOUT_ / 64), dim3(256), 0, stream,
                     concat, Wo, bo, out);
}
